// MOSoftQNetwork_75935021793657
// MI455X (gfx1250) — compile-verified, matrix operand order checked by reading, not run on hardware
//
#include <hip/hip_runtime.h>

typedef __attribute__((ext_vector_type(16))) _Float16 v16h;
typedef __attribute__((ext_vector_type(8)))  float    v8f;

// Problem constants (match reference)
#define NN 100000   // nodes  (divisible by 16 -> no row masking in GNN GEMM)
#define EE 1600000  // edges
#define DD 32       // hidden dim
#define LL 3        // layers
#define BB 64       // graphs
#define AA 16       // action dim
#define HH 256      // MLP hidden
#define RR 4        // reward dim

// ---------------------------------------------------------------------------
// WMMA fragment mapping (wave32, V_WMMA_F32_16X16X32_F16, ISA 7.12.2)
//   A (16x32 f16): lane l holds row m=l&15; half j at k=(j&7)+8*(2*(j>>3)+g)
//     -> per lane two contiguous 8-float runs of the f32 source row.
//   B (32x16 f16): lane l holds col n=nBase+(l&15); same k mapping per half.
//   C/D (16x16 f32): VGPR r <-> row r+8*g, col lane&15.
// B fragments are pre-swizzled once per launch into a packed image:
//   frags[fragIdx*32 + lane] = v16h  (so the hot loop does 2x b128 per frag).
// ---------------------------------------------------------------------------
__device__ __forceinline__ v16h load_a_tile(const float* __restrict__ X, int ld,
                                            int rowBase, int kBase, int K,
                                            float scale, int lane) {
  const int m = lane & 15;
  const int g = (lane >> 4) & 1;
  const float* rowp = X + (size_t)(rowBase + m) * (size_t)ld;
  v16h a;
#pragma unroll
  for (int j = 0; j < 16; ++j) {
    const int k = kBase + (j & 7) + 8 * (2 * (j >> 3) + g);
    const float v = (k < K) ? rowp[k] * scale : 0.0f;
    a[j] = (_Float16)v;
  }
  return a;
}

// Pre-swizzle W[K,Nc] (f32, row-major) into WMMA B fragments (f16).
// fragIdx = kTile*nTiles + nTile; one thread per (fragIdx, lane).
__global__ void __launch_bounds__(256)
pack_b_frags_kernel(const float* __restrict__ W, int K, int Nc, int nTiles,
                    int kTiles, v16h* __restrict__ frags) {
  int t = blockIdx.x * blockDim.x + threadIdx.x;
  const int total = kTiles * nTiles * 32;
  if (t >= total) return;
  const int lane = t & 31;
  const int fragIdx = t >> 5;
  const int kTile = fragIdx / nTiles;
  const int nTile = fragIdx % nTiles;
  const int n = nTile * 16 + (lane & 15);
  const int g = (lane >> 4) & 1;
  const bool nin = (n < Nc);
  v16h b;
#pragma unroll
  for (int j = 0; j < 16; ++j) {
    const int k = kTile * 32 + (j & 7) + 8 * (2 * (j >> 3) + g);
    const float v = (nin && k < K) ? W[(size_t)k * (size_t)Nc + n] : 0.0f;
    b[j] = (_Float16)v;
  }
  frags[fragIdx * 32 + lane] = b;
}

// ---------------------------------------------------------------------------
// Small utility kernels
// ---------------------------------------------------------------------------
__global__ void zero_f32_kernel(float* __restrict__ p, int n) {
  int t = blockIdx.x * blockDim.x + threadIdx.x;
  if (t < n) p[t] = 0.0f;
}

__global__ void deg_kernel(const int* __restrict__ dst, float* __restrict__ deg,
                           int e) {
  int t = blockIdx.x * blockDim.x + threadIdx.x;
  if (t < e) atomicAdd(&deg[dst[t]], 1.0f);
}

__global__ void gcnt_kernel(const int* __restrict__ gid, float* __restrict__ gcnt,
                            int n) {
  int t = blockIdx.x * blockDim.x + threadIdx.x;
  if (t < n) atomicAdd(&gcnt[gid[t]], 1.0f);
}

__global__ void invdeg_kernel(float* __restrict__ deg, int n) {
  int t = blockIdx.x * blockDim.x + threadIdx.x;
  if (t < n) deg[t] = 1.0f / fmaxf(deg[t], 1.0f);
}

// Neighbor-sum scatter: segment_sum(h[src], dst).  8 threads/edge, float4 each.
__global__ void __launch_bounds__(256)
scatter_edges_kernel(const float* __restrict__ h, const int* __restrict__ src,
                     const int* __restrict__ dst, float* __restrict__ agg,
                     int nWork) {
  int t = blockIdx.x * blockDim.x + threadIdx.x;
  if (t >= nWork) return;
  const int e = t >> 3;
  const int c = (t & 7) * 4;
  const int s = src[e];
  const int d = dst[e];
  const float4 v = *reinterpret_cast<const float4*>(h + (size_t)s * DD + c);
  float* ap = agg + (size_t)d * DD + c;
  atomicAdd(ap + 0, v.x);
  atomicAdd(ap + 1, v.y);
  atomicAdd(ap + 2, v.z);
  atomicAdd(ap + 3, v.w);
}

// ---------------------------------------------------------------------------
// GNN dense update: h_out = relu(h_in @ Wself + (agg/deg) @ Wnb + bias)
// One wave per 16-row tile; 1/deg folded into the A(agg) f16 conversion;
// bias pre-loaded into the f32 accumulator; 4 WMMAs per wave.
// B operands come from pre-swizzled fragments (2 per matrix: ntile 0/1).
// Requires nRows % 16 == 0 (NN = 6250*16).
// ---------------------------------------------------------------------------
__global__ void __launch_bounds__(256)
gnn_layer_kernel(const float* __restrict__ h_in, const float* __restrict__ agg,
                 const float* __restrict__ invdeg,
                 const v16h* __restrict__ fragSelf,
                 const v16h* __restrict__ fragNb,
                 const float* __restrict__ bias, float* __restrict__ h_out,
                 int nRows) {
  const int wave = (blockIdx.x * blockDim.x + threadIdx.x) >> 5;
  const int lane = threadIdx.x & 31;
  const int rowBase = wave * 16;
  if (rowBase >= nRows) return;  // wave-uniform: EXEC stays all-ones for WMMA

  const int m = lane & 15;
  const int g = (lane >> 4) & 1;
  const float invd = invdeg[rowBase + m];

  const v16h ah = load_a_tile(h_in, DD, rowBase, 0, DD, 1.0f, lane);
  const v16h ag = load_a_tile(agg, DD, rowBase, 0, DD, invd, lane);

#pragma unroll
  for (int t = 0; t < 2; ++t) {  // two 16-wide column tiles of the 32-wide output
    const v16h bs = fragSelf[t * 32 + lane];  // 2x global_load_b128, coalesced
    const v16h bn = fragNb[t * 32 + lane];
    const float bb = bias[16 * t + m];        // column bias
    v8f c;
#pragma unroll
    for (int r = 0; r < 8; ++r) c[r] = bb;
    c = __builtin_amdgcn_wmma_f32_16x16x32_f16(false, ag, false, bn, (short)0, c,
                                               false, false);
    c = __builtin_amdgcn_wmma_f32_16x16x32_f16(false, ah, false, bs, (short)0, c,
                                               false, false);
#pragma unroll
    for (int r = 0; r < 8; ++r) {
      const float v = fmaxf(c[r], 0.0f);
      h_out[(size_t)(rowBase + r + 8 * g) * DD + 16 * t + m] = v;
    }
  }
}

// Per-graph sum readout: gsum[gid[i]][c] += h[i][c]
__global__ void __launch_bounds__(256)
readout_kernel(const float* __restrict__ h, const int* __restrict__ gid,
               float* __restrict__ gsum, int nWork) {
  int t = blockIdx.x * blockDim.x + threadIdx.x;
  if (t >= nWork) return;
  const int i = t >> 5;
  const int c = t & 31;
  atomicAdd(&gsum[(size_t)gid[i] * DD + c], h[(size_t)i * DD + c]);
}

// x0 = concat(gsum/cnt, actions)  -> [B, D+A]
__global__ void build_x_kernel(const float* __restrict__ gsum,
                               const float* __restrict__ gcnt,
                               const float* __restrict__ actions,
                               float* __restrict__ x0) {
  int t = blockIdx.x * blockDim.x + threadIdx.x;
  if (t >= BB * (DD + AA)) return;
  const int bi = t / (DD + AA);
  const int c = t % (DD + AA);
  float v;
  if (c < DD)
    v = gsum[(size_t)bi * DD + c] / fmaxf(gcnt[bi], 1.0f);
  else
    v = actions[(size_t)bi * AA + (c - DD)];
  x0[t] = v;
}

// ---------------------------------------------------------------------------
// Generic WMMA GEMM for the MLP head: Y = act(X[M,K] @ W[K,Nc] + bias)
// One wave per 16x16 output tile; K zero-padded to a multiple of 32 (padding
// baked into the packed B fragments and the A loader).
// Requires M % 16 == 0 (M = 64).
// ---------------------------------------------------------------------------
__global__ void __launch_bounds__(256)
mlp_gemm_kernel(const float* __restrict__ X, const v16h* __restrict__ fragsW,
                const float* __restrict__ bias, float* __restrict__ Y, int M,
                int K, int Nc, int nTiles, int applyRelu) {
  const int wave = (blockIdx.x * blockDim.x + threadIdx.x) >> 5;
  const int lane = threadIdx.x & 31;
  const int mTile = wave / nTiles;
  const int nTile = wave % nTiles;
  if (mTile * 16 >= M) return;  // wave-uniform

  const int m = lane & 15;
  const int g = (lane >> 4) & 1;
  const int n = nTile * 16 + m;

  const float bb = (n < Nc) ? bias[n] : 0.0f;
  v8f c;
#pragma unroll
  for (int r = 0; r < 8; ++r) c[r] = bb;

  for (int kb = 0; kb < K; kb += 32) {
    const v16h a = load_a_tile(X, K, mTile * 16, kb, K, 1.0f, lane);
    const v16h b = fragsW[(size_t)((kb >> 5) * nTiles + nTile) * 32 + lane];
    c = __builtin_amdgcn_wmma_f32_16x16x32_f16(false, a, false, b, (short)0, c,
                                               false, false);
  }

#pragma unroll
  for (int r = 0; r < 8; ++r) {
    if (n < Nc) {
      float v = c[r];
      if (applyRelu) v = fmaxf(v, 0.0f);
      Y[(size_t)(mTile * 16 + r + 8 * g) * Nc + n] = v;
    }
  }
}

// ---------------------------------------------------------------------------
// Host-side orchestration
// ---------------------------------------------------------------------------
static inline int cdiv(long long a, int b) { return (int)((a + b - 1) / b); }

extern "C" void kernel_launch(void* const* d_in, const int* in_sizes, int n_in,
                              void* d_out, int out_size, void* d_ws,
                              size_t ws_size, hipStream_t stream) {
  (void)in_sizes; (void)n_in; (void)out_size; (void)ws_size;

  const float* nf      = (const float*)d_in[0];
  const int*   src     = (const int*)d_in[1];
  const int*   dst     = (const int*)d_in[2];
  const int*   gid     = (const int*)d_in[3];
  const float* actions = (const float*)d_in[4];
  const float* w_self  = (const float*)d_in[5];  // [L,32,32]
  const float* w_nb    = (const float*)d_in[6];  // [L,32,32]
  const float* bvec    = (const float*)d_in[7];  // [L,32]
  const float* w1      = (const float*)d_in[8];  // [48,256]
  const float* b1      = (const float*)d_in[9];
  const float* w2      = (const float*)d_in[10]; // [256,256]
  const float* b2      = (const float*)d_in[11];
  const float* w3      = (const float*)d_in[12]; // [256,4]
  const float* b3      = (const float*)d_in[13];
  float*       out     = (float*)d_out;          // [64,4]

  // Workspace layout (floats first, then 32B-aligned packed f16 fragments).
  float* ws     = (float*)d_ws;
  float* invdeg = ws;                            // [N]
  float* gcnt   = invdeg + NN;                   // [B]
  float* gsum   = gcnt + BB;                     // [B*32]
  float* hA     = gsum + (size_t)BB * DD;        // [N*32]
  float* hB     = hA + (size_t)NN * DD;          // [N*32]
  float* agg    = hB + (size_t)NN * DD;          // [N*32]
  float* x0     = agg + (size_t)NN * DD;         // [B*48]
  float* x1     = x0 + (size_t)BB * (DD + AA);   // [B*256]
  float* x2     = x1 + (size_t)BB * HH;          // [B*256]
  float* fend   = x2 + (size_t)BB * HH;
  // Round up to 32B (8-float) boundary for v16h fragments.
  size_t foff = (size_t)(fend - ws);
  foff = (foff + 7) & ~(size_t)7;
  v16h* fragGnn = (v16h*)(ws + foff);            // 6 matrices * 2 frags
  v16h* fragW1  = fragGnn + (size_t)LL * 2 * 2 * 32;  // kTiles=2, nTiles=16
  v16h* fragW2  = fragW1 + (size_t)2 * 16 * 32;       // kTiles=8, nTiles=16
  v16h* fragW3  = fragW2 + (size_t)8 * 16 * 32;       // kTiles=8, nTiles=1

  const int TB = 256;

  // Pre-swizzle all weight matrices into WMMA B fragments (f16).
  for (int l = 0; l < LL; ++l) {
    pack_b_frags_kernel<<<1, 64, 0, stream>>>(
        w_self + (size_t)l * DD * DD, DD, DD, 2, 1, fragGnn + (size_t)(2 * l) * 2 * 32);
    pack_b_frags_kernel<<<1, 64, 0, stream>>>(
        w_nb + (size_t)l * DD * DD, DD, DD, 2, 1, fragGnn + (size_t)(2 * l + 1) * 2 * 32);
  }
  pack_b_frags_kernel<<<cdiv(2 * 16 * 32, TB), TB, 0, stream>>>(
      w1, DD + AA, HH, 16, 2, fragW1);
  pack_b_frags_kernel<<<cdiv(8 * 16 * 32, TB), TB, 0, stream>>>(
      w2, HH, HH, 16, 8, fragW2);
  pack_b_frags_kernel<<<1, 8 * 1 * 32, 0, stream>>>(
      w3, HH, RR, 1, 8, fragW3);

  // Zero all atomic accumulators (invdeg|gcnt|gsum are contiguous).
  {
    const int n = NN + BB + BB * DD;
    zero_f32_kernel<<<cdiv(n, TB), TB, 0, stream>>>(invdeg, n);
  }
  // Degrees and per-graph node counts.
  deg_kernel<<<cdiv(EE, TB), TB, 0, stream>>>(dst, invdeg, EE);
  gcnt_kernel<<<cdiv(NN, TB), TB, 0, stream>>>(gid, gcnt, NN);
  invdeg_kernel<<<cdiv(NN, TB), TB, 0, stream>>>(invdeg, NN);

  // GNN layers: nf -> hA -> hB -> hA
  const float* hin = nf;
  float* hout = hA;
  for (int l = 0; l < LL; ++l) {
    zero_f32_kernel<<<cdiv((long long)NN * DD, TB), TB, 0, stream>>>(
        agg, NN * DD);
    scatter_edges_kernel<<<cdiv((long long)EE * 8, TB), TB, 0, stream>>>(
        hin, src, dst, agg, EE * 8);
    const int waves = NN / 16;  // 6250
    gnn_layer_kernel<<<cdiv((long long)waves * 32, TB), TB, 0, stream>>>(
        hin, agg, invdeg, fragGnn + (size_t)(2 * l) * 2 * 32,
        fragGnn + (size_t)(2 * l + 1) * 2 * 32, bvec + (size_t)l * DD, hout, NN);
    hin = hout;
    hout = (hout == hA) ? hB : hA;
  }
  const float* hfin = hin;  // hA after 3 layers

  // Per-graph mean readout + concat with actions.
  readout_kernel<<<cdiv((long long)NN * DD, TB), TB, 0, stream>>>(
      hfin, gid, gsum, NN * DD);
  build_x_kernel<<<cdiv(BB * (DD + AA), TB), TB, 0, stream>>>(gsum, gcnt,
                                                              actions, x0);

  // MLP head (WMMA).
  {
    const int nT1 = HH / 16;  // 16
    mlp_gemm_kernel<<<cdiv((long long)(BB / 16) * nT1 * 32, TB), TB, 0, stream>>>(
        x0, fragW1, b1, x1, BB, DD + AA, HH, nT1, 1);
    mlp_gemm_kernel<<<cdiv((long long)(BB / 16) * nT1 * 32, TB), TB, 0, stream>>>(
        x1, fragW2, b2, x2, BB, HH, HH, nT1, 1);
    const int nT3 = 1;  // ceil(4/16)
    mlp_gemm_kernel<<<cdiv((long long)(BB / 16) * nT3 * 32, TB), TB, 0, stream>>>(
        x2, fragW3, b3, out, BB, HH, RR, nT3, 0);
  }
}
